// TreeFCN_42021960024434
// MI455X (gfx1250) — matrix operand order repaired and checked by reading, not compiled, not run on hardware
//
#include <hip/hip_runtime.h>
#include <hip/hip_bf16.h>
#include <stdint.h>

// ---------------------------------------------------------------------------
// TreeFCN head for MI455X (gfx1250, wave32).
//  * All convs run as bf16 WMMA GEMMs (v_wmma_f32_16x16x32_bf16, f32 accum);
//    each wave computes a 16(out-ch) x 32(pixel) D strip: one A (weight)
//    fragment feeds two back-to-back WMMAs per K-step.
//  * MST + BFS-rooting + grouped tree filter fused into ONE workgroup per
//    batch image (1024 threads = 1/pixel) living in LDS (~100 KB of 320 KB),
//    ds_min_u64 atomics for Boruvka edge election, workgroup barriers for the
//    level-synchronous sweeps. Deterministic (no float atomics).
//  * Edge-weight table is staged to LDS once via the Tensor Data Mover
//    (tensor_load_to_lds + s_wait_tensorcnt), reused for all 12 Boruvka rounds.
// ---------------------------------------------------------------------------

typedef __attribute__((ext_vector_type(16))) __bf16 v16bf;
typedef __attribute__((ext_vector_type(8)))  float  v8f;

#define HW    32
#define NPIX  1024
#define NEDGE 1984   // 32*31 horizontal + 31*32 vertical

enum { EP_BN = 1, EP_RELU = 2, EP_RES = 4 };

__device__ __forceinline__ __bf16 f2bf(float f) {
  uint32_t u = __builtin_bit_cast(uint32_t, f);
  u += 0x7fffu + ((u >> 16) & 1u);            // round-to-nearest-even
  uint16_t h = (uint16_t)(u >> 16);
  return __builtin_bit_cast(__bf16, h);
}

__device__ __forceinline__ void edge_uv(int e, int& u, int& v) {
  if (e < 992) { int r = e / 31; int c = e - r * 31; u = r * 32 + c; v = u + 1; }
  else         { u = e - 992; v = u + 32; }
}

// ---------------------------------------------------------------------------
// Conv-as-GEMM with WMMA. X:[B][Cin][1024], W:[Cout][Cin(*9)], Out:[B][Cout][1024].
// One wave per 16(out-ch) x 32(pixel) strip (2 accumulators, shared A).
// Fragment packing per CDNA5 ISA 7.12.2 wave32 layouts:
//   A (16x32 bf16): M = lane&15 ; elem i -> K = (i<8 ? i : i+8) + 8*laneHi
//                   (K-pairs contiguous -> float2 weight loads)
//   B (32x16 bf16): N = lane&15 ; elem i -> K = (i<8 ? i : i+8) + 8*laneHi (same as A)
//   D (16x16 f32) : VGPR r -> M = r + 8*laneHi ; N = lane&15
// ---------------------------------------------------------------------------
__global__ void k_conv_wmma(const float* __restrict__ X, const float* __restrict__ W,
                            float* __restrict__ Out, int Cin, int Cout, int is3x3, int mode,
                            const float* __restrict__ g, const float* __restrict__ bb,
                            const float* __restrict__ m, const float* __restrict__ vv,
                            const float* __restrict__ resid)
{
  const int lane  = threadIdx.x & 31;
  const int hi    = lane >> 4;
  const int ln    = lane & 15;
  const int tileP = blockIdx.x * 32;       // 32-pixel strip (one image row)
  const int co0   = blockIdx.y * 16;       // out-channel tile
  const int bz    = blockIdx.z;
  const int K     = is3x3 ? Cin * 9 : Cin;
  const float* Xb = X + (size_t)bz * Cin * NPIX;

  const int px0 = tileP + ln;              // B-column pixels for the two tiles
  const int px1 = tileP + 16 + ln;
  const int py0 = px0 >> 5, pxx0 = px0 & 31;
  const int py1 = px1 >> 5, pxx1 = px1 & 31;
  const int am  = co0 + ln;                // A-row (out channel) for this lane

  v8f acc0 = {}, acc1 = {};
  for (int k0 = 0; k0 < K; k0 += 32) {
    v16bf A, B0, B1;
    const float* Wr = W + (size_t)am * K + k0;
#pragma unroll
    for (int i2 = 0; i2 < 8; ++i2) {       // vectorized weight loads (K-pairs)
      int kk = (i2 < 4 ? 2 * i2 : 2 * i2 + 8) + (hi ? 8 : 0);
      float2 wv2 = *(const float2*)(Wr + kk);
      A[2 * i2]     = f2bf(wv2.x);
      A[2 * i2 + 1] = f2bf(wv2.y);
    }
#pragma unroll
    for (int i = 0; i < 16; ++i) {
      int kk = k0 + (i < 8 ? i : i + 8) + (hi ? 8 : 0);
      float xv0, xv1;
      if (!is3x3) {
        const float* Xr = Xb + (size_t)kk * NPIX;
        xv0 = Xr[px0];
        xv1 = Xr[px1];
      } else {                             // on-the-fly im2col, zero pad
        int ci = kk / 9, tap = kk - ci * 9;
        int dy = tap / 3 - 1, dx = tap - (tap / 3) * 3 - 1;
        const float* Xc = Xb + (size_t)ci * NPIX;
        int yy0 = py0 + dy, xx0 = pxx0 + dx;
        int yy1 = py1 + dy, xx1 = pxx1 + dx;
        xv0 = (yy0 >= 0 && yy0 < HW && xx0 >= 0 && xx0 < HW) ? Xc[yy0 * HW + xx0] : 0.0f;
        xv1 = (yy1 >= 0 && yy1 < HW && xx1 >= 0 && xx1 < HW) ? Xc[yy1 * HW + xx1] : 0.0f;
      }
      B0[i] = f2bf(xv0);
      B1[i] = f2bf(xv1);
    }
    acc0 = __builtin_amdgcn_wmma_f32_16x16x32_bf16(false, A, false, B0,
                                                   (short)0, acc0, false, false);
    acc1 = __builtin_amdgcn_wmma_f32_16x16x32_bf16(false, A, false, B1,
                                                   (short)0, acc1, false, false);
  }
#pragma unroll
  for (int r = 0; r < 8; ++r) {
    int co = co0 + r + hi * 8;
    float sc = 1.0f, sh = 0.0f;
    if (mode & EP_BN) {
      sc = g[co] * rsqrtf(vv[co] + 1e-5f);
      sh = bb[co] - m[co] * sc;
    }
    size_t orow = ((size_t)bz * Cout + co) * NPIX;
    float v0 = acc0[r] * sc + sh;
    float v1 = acc1[r] * sc + sh;
    if (mode & EP_RES) { v0 += resid[orow + px0]; v1 += resid[orow + px1]; }
    if (mode & EP_RELU) { v0 = fmaxf(v0, 0.0f); v1 = fmaxf(v1, 0.0f); }
    Out[orow + px0] = v0;
    Out[orow + px1] = v1;
  }
}

// ---------------------------------------------------------------------------
// Edge weights: dist[b][e] = sum_c (fm[c][u]-fm[c][v])^2 over 1024 channels.
// ---------------------------------------------------------------------------
__global__ void k_edge_dist(const float* __restrict__ fm, float* __restrict__ dist) {
  int e  = blockIdx.x * blockDim.x + threadIdx.x;
  int bz = blockIdx.y;
  if (e >= NEDGE) return;
  int u, v; edge_uv(e, u, v);
  const float* f = fm + (size_t)bz * 1024 * NPIX;
  float s = 0.0f;
  for (int c = 0; c < 1024; ++c) {
    float d = f[(size_t)c * NPIX + u] - f[(size_t)c * NPIX + v];
    s = fmaf(d, d, s);
  }
  dist[bz * NEDGE + e] = s;
}

// ---------------------------------------------------------------------------
// One workgroup per batch: Boruvka MST -> BFS root/depth -> grouped tree
// filter (two level-synchronous sweeps) -> fusion = latent + filtered/norm.
// Dynamic LDS layout (102,672 B):
//   [ f32 dist 8K | u64 key 8K | f32 agg 17x1024 (68K) | par 4K | pn 4K |
//     dep 4K | w 4K | mst bitmask 256B | misc 16B ]
// ---------------------------------------------------------------------------
#define TREE_SMEM (8192 + 8192 + 17 * 1024 * 4 + 3 * 4096 + 4096 + 64 * 4 + 16)

__global__ void __launch_bounds__(1024, 1) k_mst_treefilter(
    const float* __restrict__ dist, const float* __restrict__ feat,
    const float* __restrict__ emb,  const float* __restrict__ latent,
    float* __restrict__ fusion)
{
  const int N = 1024;
  extern __shared__ unsigned char smem[];
  float* s_dist  = (float*)smem;                                       // 8 KB
  unsigned long long* s_key = (unsigned long long*)(smem + 8192);      // 8 KB
  float* s_agg   = (float*)(smem + 16384);                             // 68 KB
  int*   s_par   = (int*)(smem + 16384 + 17 * 1024 * 4);
  int*   s_pn    = s_par + N;
  int*   s_dep   = s_pn + N;
  float* s_w     = (float*)(s_dep + N);
  unsigned* s_mst = (unsigned*)(s_w + N);                              // 64 words
  int*   s_misc  = (int*)(s_mst + 64);

  const int t  = threadIdx.x;                 // node id (pixel)
  const int bz = blockIdx.x;
  const float* db    = dist   + (size_t)bz * NEDGE;
  const float* featb = feat   + (size_t)bz * 64 * N;
  const float* embb  = emb    + (size_t)bz * 32 * N;
  const float* latb  = latent + (size_t)bz * 64 * N;
  float*       fusb  = fusion + (size_t)bz * 64 * N;

  __builtin_prefetch(&featb[t], 0, 0);        // global_prefetch_b8 (gfx1250)

  // ---- stage dist[b] into LDS once (reused for all 12 Boruvka rounds) ----
#if defined(__AMDGCN__) && __has_builtin(__builtin_amdgcn_tensor_load_to_lds)
  if (t < 32) {                               // TDM is per-wave; one wave issues
    typedef uint32_t u32x4 __attribute__((ext_vector_type(4)));
    typedef int      i32x4 __attribute__((ext_vector_type(4)));
    typedef int      i32x8 __attribute__((ext_vector_type(8)));
    uint64_t ga = (uint64_t)(uintptr_t)db;
    uint32_t la = (uint32_t)(uintptr_t)s_dist;            // LDS byte address
    u32x4 g0;
    g0[0] = 1u;                                           // count=1 (user D#)
    g0[1] = la;                                           // lds_addr
    g0[2] = (uint32_t)ga;                                 // global_addr[31:0]
    g0[3] = (uint32_t)((ga >> 32) & 0x01ffffffu) | (2u << 30);  // [56:32]|type=2
    i32x8 g1;
    g1[0] = (int)(2u << 16);                              // data_size = 4 B
    g1[1] = (int)((uint32_t)NEDGE << 16);                 // tensor_dim0 = 1984
    g1[2] = (int)(1u << 16);                              // tensor_dim1 = 1
    g1[3] = (int)((uint32_t)NEDGE << 16);                 // tile_dim0 = 1984
    g1[4] = 1;                                            // tile_dim1 = 1
    g1[5] = NEDGE;                                        // dim0 stride (1 row)
    g1[6] = 0;
    g1[7] = 0;
    i32x4 z4 = {0, 0, 0, 0};
#if defined(__clang_major__) && (__clang_major__ >= 23)
    i32x8 z8 = {0, 0, 0, 0, 0, 0, 0, 0};
    __builtin_amdgcn_tensor_load_to_lds(g0, g1, z4, z4, z8, 0);
#else
    __builtin_amdgcn_tensor_load_to_lds(g0, g1, z4, z4, 0);
#endif
    __builtin_amdgcn_s_wait_tensorcnt(0);
  }
  __syncthreads();
#else
  for (int e = t; e < NEDGE; e += N) s_dist[e] = db[e];
  __syncthreads();
#endif

  // ---------------- Boruvka MST (union-find in LDS) ----------------
  s_par[t] = t;
  if (t < 64) s_mst[t] = 0u;
  __syncthreads();
  for (int round = 0; round < 12; ++round) {
    s_key[t] = ~0ull;
    __syncthreads();
    for (int e = t; e < NEDGE; e += N) {      // elect min edge per component
      int u, v; edge_uv(e, u, v);
      int cu = s_par[u], cv = s_par[v];
      if (cu != cv) {
        unsigned long long key =
            ((unsigned long long)__float_as_uint(s_dist[e]) << 32) | (unsigned)e;
        atomicMin(&s_key[cu], key);           // ds_min_u64
        atomicMin(&s_key[cv], key);
      }
    }
    __syncthreads();
    int prop = s_par[t];
    if (s_par[t] == t && s_key[t] != ~0ull) { // roots adopt their min edge
      int e = (int)(s_key[t] & 0xffffffffu);
      int u, v; edge_uv(e, u, v);
      atomicOr(&s_mst[e >> 5], 1u << (e & 31));
      int cu = s_par[u], cv = s_par[v];
      prop = (cu == t) ? cv : cu;
    }
    s_pn[t] = prop;                           // scratch: proposed parents
    __syncthreads();
    int p  = s_pn[t];
    int pp = s_pn[p];
    s_par[t] = (pp == t && t < p) ? t : p;    // break mutual 2-cycles
    __syncthreads();
    for (int it = 0; it < 10; ++it) {         // flatten by pointer doubling
      int a  = s_par[t];
      int gg = s_par[a];
      __syncthreads();
      s_par[t] = gg;
      __syncthreads();
    }
  }

  // ---------------- BFS rooting at node 0 + depth ----------------
  int* s_vis = (int*)s_key;                   // alias: Boruvka keys done
  s_pn[t]  = (t == 0) ? 0 : -1;
  s_dep[t] = 0;
  __syncthreads();
  for (int lev = 1; lev < N; ++lev) {
    s_vis[t] = (s_pn[t] >= 0) ? 1 : 0;
    __syncthreads();
    if (t == 0) s_misc[0] = 0;
    __syncthreads();
    for (int e = t; e < NEDGE; e += N) {
      if ((s_mst[e >> 5] >> (e & 31)) & 1u) {
        int u, v; edge_uv(e, u, v);
        int vu = s_vis[u], vv2 = s_vis[v];
        if (vu && !vv2)       { s_pn[v] = u; s_dep[v] = lev; s_misc[0] = 1; }
        else if (vv2 && !vu)  { s_pn[u] = v; s_dep[u] = lev; s_misc[0] = 1; }
      }
    }
    __syncthreads();
    if (s_misc[0] == 0) break;
    __syncthreads();
  }
  if (t == 0) s_misc[1] = 0;
  __syncthreads();
  atomicMax(&s_misc[1], s_dep[t]);
  __syncthreads();
  const int maxd = s_misc[1];
  const int pnv  = s_pn[t];
  const int depv = s_dep[t];

  // grid neighbors of this node (candidate children)
  int y = t >> 5, x = t & 31;
  int nb[4]; int nn = 0;
  if (x > 0)  nb[nn++] = t - 1;
  if (x < 31) nb[nn++] = t + 1;
  if (y > 0)  nb[nn++] = t - 32;
  if (y < 31) nb[nn++] = t + 32;

  // ---------------- grouped tree filter (4 groups x 17 ch) ----------------
  for (int g = 0; g < 4; ++g) {
    float sum = 0.0f;
#pragma unroll
    for (int c = 0; c < 8; ++c) {
      const float* ec = embb + (size_t)(g * 8 + c) * N;
      float d = ec[t] - ec[pnv];
      sum = fmaf(d, d, sum);
    }
    s_w[t] = (t == 0) ? 0.0f : expf(-sum);    // edge weight to parent
#pragma unroll
    for (int c = 0; c < 16; ++c)
      s_agg[c * N + t] = featb[(size_t)(g * 16 + c) * N + t];
    s_agg[16 * N + t] = 1.0f;                 // normalization channel
    __syncthreads();

    // up sweep (leaves -> root): pull from children; deterministic
    for (int lev = maxd - 1; lev >= 0; --lev) {
      if (depv == lev) {
        for (int j = 0; j < nn; ++j) {
          int u = nb[j];
          if (s_pn[u] == t) {                 // u is a tree child of t
            float wu = s_w[u];
#pragma unroll
            for (int c = 0; c < 17; ++c)
              s_agg[c * N + t] += wu * s_agg[c * N + u];
          }
        }
      }
      __syncthreads();
    }
    // down sweep (root -> leaves), in place
    for (int lev = 1; lev <= maxd; ++lev) {
      if (depv == lev) {
        float wv = s_w[t];
#pragma unroll
        for (int c = 0; c < 17; ++c) {
          float av = s_agg[c * N + t];
          s_agg[c * N + t] = fmaf(wv, s_agg[c * N + pnv] - wv * av, av);
        }
      }
      __syncthreads();
    }
    float norm = s_agg[16 * N + t];
#pragma unroll
    for (int c = 0; c < 16; ++c) {
      size_t oi = (size_t)(g * 16 + c) * N + t;
      fusb[oi] = latb[oi] + s_agg[c * N + t] / norm;
    }
    __syncthreads();
  }
}

// ---------------------------------------------------------------------------
extern "C" void kernel_launch(void* const* d_in, const int* in_sizes, int n_in,
                              void* d_out, int out_size, void* d_ws, size_t ws_size,
                              hipStream_t stream) {
  (void)in_sizes; (void)n_in; (void)out_size; (void)ws_size;
  const float* fm      = (const float*)d_in[0];
  const float* last_fm = (const float*)d_in[1];
  const float* lat_w1  = (const float*)d_in[2];
  const float* lat_w3a = (const float*)d_in[3];
  const float* lat_g   = (const float*)d_in[4];
  const float* lat_b   = (const float*)d_in[5];
  const float* lat_m   = (const float*)d_in[6];
  const float* lat_v   = (const float*)d_in[7];
  const float* lat_w3b = (const float*)d_in[8];
  const float* ref_w1  = (const float*)d_in[9];
  const float* ref_w3a = (const float*)d_in[10];
  const float* ref_g   = (const float*)d_in[11];
  const float* ref_b   = (const float*)d_in[12];
  const float* ref_m   = (const float*)d_in[13];
  const float* ref_v   = (const float*)d_in[14];
  const float* ref_w3b = (const float*)d_in[15];
  const float* emb_w   = (const float*)d_in[16];
  float* out = (float*)d_out;

  // workspace carve-up (floats); ~3.4 MB total
  float* ws = (float*)d_ws;
  const size_t SZ64 = (size_t)2 * 64 * NPIX;
  float* x1l    = ws; ws += SZ64;
  float* tl     = ws; ws += SZ64;
  float* latent = ws; ws += SZ64;
  float* embed  = ws; ws += (size_t)2 * 32 * NPIX;
  float* distb  = ws; ws += 4096;
  float* fusion = ws; ws += SZ64;
  float* x1r    = ws; ws += SZ64;
  float* tr     = ws; ws += SZ64;

  dim3 blk(32);
  // latent = RefineResidual(fm)
  k_conv_wmma<<<dim3(32, 4, 2), blk, 0, stream>>>(fm, lat_w1, x1l, 1024, 64, 0, 0,
      nullptr, nullptr, nullptr, nullptr, nullptr);
  k_conv_wmma<<<dim3(32, 4, 2), blk, 0, stream>>>(x1l, lat_w3a, tl, 64, 64, 1,
      EP_BN | EP_RELU, lat_g, lat_b, lat_m, lat_v, nullptr);
  k_conv_wmma<<<dim3(32, 4, 2), blk, 0, stream>>>(tl, lat_w3b, latent, 64, 64, 1,
      EP_RES | EP_RELU, nullptr, nullptr, nullptr, nullptr, x1l);
  // embed = 1x1(last_fm)
  k_conv_wmma<<<dim3(32, 2, 2), blk, 0, stream>>>(last_fm, emb_w, embed, 64, 32, 0, 0,
      nullptr, nullptr, nullptr, nullptr, nullptr);
  // grid-edge feature distances on fm
  k_edge_dist<<<dim3(8, 2), dim3(256), 0, stream>>>(fm, distb);
  // MST + BFS + tree filter + fusion, one workgroup per batch image
  hipFuncSetAttribute(reinterpret_cast<const void*>(k_mst_treefilter),
                      hipFuncAttributeMaxDynamicSharedMemorySize, (int)TREE_SMEM);
  k_mst_treefilter<<<dim3(2), dim3(1024), TREE_SMEM, stream>>>(
      distb, last_fm, embed, latent, fusion);
  // final RefineResidual(fusion) -> out
  k_conv_wmma<<<dim3(32, 4, 2), blk, 0, stream>>>(fusion, ref_w1, x1r, 64, 64, 0, 0,
      nullptr, nullptr, nullptr, nullptr, nullptr);
  k_conv_wmma<<<dim3(32, 4, 2), blk, 0, stream>>>(x1r, ref_w3a, tr, 64, 64, 1,
      EP_BN | EP_RELU, ref_g, ref_b, ref_m, ref_v, nullptr);
  k_conv_wmma<<<dim3(32, 4, 2), blk, 0, stream>>>(tr, ref_w3b, out, 64, 64, 1,
      EP_RES | EP_RELU, nullptr, nullptr, nullptr, nullptr, x1r);
}
